// MultiHeadAttentionXL_89326729822477
// MI455X (gfx1250) — matrix operand order repaired and checked by reading, not compiled, not run on hardware
//
#include <hip/hip_runtime.h>

typedef __bf16 bf16;
typedef __attribute__((ext_vector_type(16))) __bf16 v16bf;
typedef __attribute__((ext_vector_type(8)))  float  v8f;

#define SEQ    1024
#define BSZ    8
#define NH     16
#define DI     64
#define DM     1024   // d_input == H*d
#define KVW    2048   // 2*H*d
#define NEGINF (-3.0e38f)
#define ATT_SCALE 0.125f

// ---------------- WMMA helper (gfx1250, wave32) ----------------

__device__ inline v8f wmma_bf16(v16bf a, v16bf b, v8f c) {
  return __builtin_amdgcn_wmma_f32_16x16x32_bf16(false, a, false, b, (short)0, c,
                                                 false, false);
}

// A-fragment swizzle (16-bit A 16x32, contiguous k order):
// element (row r in 0..15, k) -> flat index inside a [2][32][16] region:
//   frag f = k>>5 ; kk = k&31
//   lane = r + 16*((kk>>4)&1) ; elem = kk&15
__device__ __host__ inline int aswz(int r, int k) {
  int f = k >> 5, kk = k & 31;
  int lane = r + (((kk >> 4) & 1) << 4);
  int e = kk & 15;
  return f * 512 + lane * 16 + e;
}

// ---------------- elementwise kernels ----------------

__global__ void f32_to_bf16_kernel(const float* __restrict__ in,
                                   bf16* __restrict__ out, int n) {
  int i = blockIdx.x * blockDim.x + threadIdx.x;
  if (i < n) out[i] = (bf16)in[i];
}

// out[n*K + k] = bf16(in[k*N + n])  (weights -> [N][K] for k-contiguous B frags)
__global__ void transpose_to_bf16_kernel(const float* __restrict__ in,
                                         bf16* __restrict__ out, int N, int K) {
  size_t idx = (size_t)blockIdx.x * blockDim.x + threadIdx.x;
  if (idx >= (size_t)N * K) return;
  int n = (int)(idx / K), k = (int)(idx % K);
  out[idx] = (bf16)in[(size_t)k * N + n];
}

// kv: (S,B,2048) f32 ; q: (S,B,1024) f32 ; p: (S,1024) f32
// qswz: A-fragment-swizzled q, per (s/16, b, h): 1024 bf16 block
// kpb : bf16(k + p) row-major (S,B,H*d)
// vt  : bf16(val) transposed per (b,h): [b][h][d][SEQ]
__global__ void fold_kernel(const float* __restrict__ kv, const float* __restrict__ q,
                            const float* __restrict__ p, bf16* __restrict__ qswz,
                            bf16* __restrict__ kpb, bf16* __restrict__ vt) {
  size_t idx = (size_t)blockIdx.x * blockDim.x + threadIdx.x;
  if (idx >= (size_t)SEQ * BSZ * DM) return;
  int c = (int)(idx & (DM - 1));
  int b = (int)((idx >> 10) & (BSZ - 1));
  int s = (int)(idx >> 13);
  int h = c >> 6, dd = c & 63;
  size_t kvo = ((size_t)s * BSZ + b) * KVW + c;
  size_t qi = ((((size_t)(s >> 4) * BSZ + b) * NH + h) << 10) + aswz(s & 15, dd);
  qswz[qi] = (bf16)q[idx];
  kpb[idx] = (bf16)(kv[kvo] + p[(size_t)s * DM + c]);
  vt[(((size_t)b * NH + h) * DI + dd) * SEQ + s] = (bf16)kv[kvo + DM];
}

// bias[j,b,h] = u[h]·k[j,b,h] + v[h]·p[j,h]   (layout j*128 + b*16 + h)
__global__ void bias_kernel(const float* __restrict__ kv, const float* __restrict__ p,
                            const float* __restrict__ u, const float* __restrict__ v,
                            float* __restrict__ bias) {
  int idx = blockIdx.x * blockDim.x + threadIdx.x;
  if (idx >= SEQ * BSZ * NH) return;
  int h = idx & (NH - 1);
  int b = (idx >> 4) & (BSZ - 1);
  int j = idx >> 7;
  const float* kr = kv + ((size_t)j * BSZ + b) * KVW + h * DI;
  const float* pr = p + (size_t)j * DM + h * DI;
  const float* ur = u + h * DI;
  const float* vr = v + h * DI;
  float acc = 0.f;
#pragma unroll 8
  for (int d = 0; d < DI; ++d) acc += ur[d] * kr[d] + vr[d] * pr[d];
  bias[idx] = acc;
}

// ---------------- tiled bf16 GEMM: C(f32) = A(MxK) * Bt(NxK)^T ----------------
// 64x64 C tile / WG, 4 waves, BK=64, double-buffered LDS (one barrier per
// K-step). Next tile's staging is issued right after the WMMA chain so the
// global loads drain under the matrix pipe; staging registers are short-lived
// (no spills). Every WMMA operand is one 32B v16bf load (2x ds_load_b128).

__global__ __launch_bounds__(128) void gemm_bf16_kernel(
    const bf16* __restrict__ A, const bf16* __restrict__ Bt, float* __restrict__ C,
    int M, int N, int K) {
  __shared__ __align__(32) bf16 As[2][4][2][32][16];  // [buf][m-sub][kchunk][lane][e]
  __shared__ __align__(32) bf16 Bs[2][4][2][32][16];
  const int tid  = threadIdx.x;
  const int lane = tid & 31;
  const int wave = tid >> 5;
  const int n0 = blockIdx.x * 64;
  const int m0 = blockIdx.y * 64;

  // per-thread staging coordinates (4 float4 chunks each for A and B)
  const int srow = tid >> 3;                 // 0..15 (+16*c)
  const int sk0  = (tid & 7) * 8;            // 0,8,...,56
  const int skk  = sk0 & 31;
  const int sf   = sk0 >> 5;
  const int slnA = (srow & 15) + (((skk >> 3) & 1) << 4);
  const int sebA = ((skk >> 4) & 1) << 3;
  const int slnB = (srow & 15) + (((skk >> 3) & 1) << 4);
  const int sebB = ((skk >> 4) & 1) << 3;

  auto stage = [&](int kb0, int buf) {
#pragma unroll
    for (int c = 0; c < 4; ++c) {  // 16 rows apart per chunk; row&15 invariant
      int r = srow + 16 * c;
      float4 da = *reinterpret_cast<const float4*>(A + (size_t)(m0 + r) * K + kb0 + sk0);
      float4 db = *reinterpret_cast<const float4*>(Bt + (size_t)(n0 + r) * K + kb0 + sk0);
      *reinterpret_cast<float4*>(&As[buf][r >> 4][sf][slnA][sebA]) = da;
      *reinterpret_cast<float4*>(&Bs[buf][r >> 4][sf][slnB][sebB]) = db;
    }
  };

  v8f acc[4] = {};
  const int nit = K >> 6;
  stage(0, 0);
  __syncthreads();
  for (int i = 0; i < nit; ++i) {
    const int buf = i & 1;
    if (i + 1 < nit) {  // warm L2/L0 for the next tile while this one computes
      __builtin_prefetch(A + (size_t)(m0 + srow) * K + ((i + 1) << 6) + sk0, 0, 1);
      __builtin_prefetch(Bt + (size_t)(n0 + srow) * K + ((i + 1) << 6) + sk0, 0, 1);
    }
    v16bf af0 = *reinterpret_cast<const v16bf*>(&As[buf][wave][0][lane][0]);
    v16bf af1 = *reinterpret_cast<const v16bf*>(&As[buf][wave][1][lane][0]);
#pragma unroll
    for (int nt = 0; nt < 4; ++nt) {
      v16bf b0 = *reinterpret_cast<const v16bf*>(&Bs[buf][nt][0][lane][0]);
      v16bf b1 = *reinterpret_cast<const v16bf*>(&Bs[buf][nt][1][lane][0]);
      acc[nt] = wmma_bf16(af0, b0, acc[nt]);
      acc[nt] = wmma_bf16(af1, b1, acc[nt]);
    }
    if (i + 1 < nit) {
      stage((i + 1) << 6, 1 - buf);  // issues while WMMA pipe drains
      __syncthreads();
    }
  }
  const int rbase = m0 + wave * 16 + ((lane >> 4) << 3);
  const int cbase = n0 + (lane & 15);
#pragma unroll
  for (int nt = 0; nt < 4; ++nt)
#pragma unroll
    for (int vv = 0; vv < 8; ++vv)
      C[(size_t)(rbase + vv) * N + cbase + nt * 16] = acc[nt][vv];
}

// ---------------- flash attention per (b,h), 64-row i-block ----------------

__global__ __launch_bounds__(128) void attn_kernel(
    const bf16* __restrict__ Qs, const bf16* __restrict__ Kp,
    const bf16* __restrict__ Vt, const float* __restrict__ bias,
    bf16* __restrict__ O) {
  __shared__ __align__(32) bf16 Ks[64 * 64];       // [j][d]  (row-major)
  __shared__ __align__(32) bf16 Vs[64 * 64];       // [d][j]  (transposed)
  __shared__ float biasS[64];
  __shared__ __align__(32) bf16 Ps[4][1024];       // per-wave P, A-frag layout
  const int tid  = threadIdx.x;
  const int lane = tid & 31;
  const int wave = tid >> 5;
  const int bh = blockIdx.y;
  const int b  = bh >> 4;
  const int h  = bh & 15;
  const int i0 = blockIdx.x * 64;
  const int coff = h * DI;
  const int hi8  = (lane >> 4) << 3;
  const int ncol = lane & 15;
  const int kb   = (lane < 16) ? 0 : 16;

  // Q fragments: pre-swizzled in global, one 32B load each
  const bf16* qbase =
      Qs + ((((size_t)(i0 >> 4) + wave) * BSZ + b) * NH + h) * 1024;
  v16bf aq0 = *reinterpret_cast<const v16bf*>(qbase + lane * 16);
  v16bf aq1 = *reinterpret_cast<const v16bf*>(qbase + 512 + lane * 16);

  const bf16* vtb = Vt + ((size_t)b * NH + h) * DI * SEQ;  // rows d, len SEQ

  float m_[8], l_[8];
  v8f o[4] = {};
#pragma unroll
  for (int vv = 0; vv < 8; ++vv) { m_[vv] = NEGINF; l_[vv] = 0.f; }

  const int njb = blockIdx.x + 1;  // causal: j-blocks 0..ib
  for (int jb = 0; jb < njb; ++jb) {
    const int j0 = jb * 64;
#pragma unroll
    for (int c = 0; c < 4; ++c) {  // stage K' [j][d] and V^T [d][j] tiles
      int idx = tid + 128 * c;     // 512 float4 chunks
      int row = idx >> 3, col = (idx & 7) * 8;
      *reinterpret_cast<float4*>(&Ks[row * 64 + col]) =
          *reinterpret_cast<const float4*>(
              Kp + ((size_t)(j0 + row) * BSZ + b) * DM + coff + col);
      *reinterpret_cast<float4*>(&Vs[row * 64 + col]) =
          *reinterpret_cast<const float4*>(vtb + (size_t)row * SEQ + j0 + col);
    }
    if (tid < 64) biasS[tid] = bias[(size_t)(j0 + tid) * (BSZ * NH) + b * NH + h];
    if (jb + 1 < njb) {  // prefetch next j-block (global_prefetch_b8)
      int jn = j0 + 64;
      __builtin_prefetch(Kp + ((size_t)(jn + (tid >> 1)) * BSZ + b) * DM + coff, 0, 1);
      __builtin_prefetch(vtb + (size_t)(tid >> 1) * SEQ + jn, 0, 1);
    }
    __syncthreads();

    // scores: B(k=d, n=j) = Ks[j][d] -> per-lane 16 contiguous d values
    v8f st[4];
#pragma unroll
    for (int jt = 0; jt < 4; ++jt) {
      v8f s = {};
      const bf16* krow = &Ks[(jt * 16 + ncol) * 64 + kb];
      s = wmma_bf16(aq0, *reinterpret_cast<const v16bf*>(krow), s);
      s = wmma_bf16(aq1, *reinterpret_cast<const v16bf*>(krow + 32), s);
      st[jt] = s;
    }
    const int rrow = i0 + wave * 16 + hi8;  // + vv = global query row
    float rowmax[8];
#pragma unroll
    for (int vv = 0; vv < 8; ++vv) rowmax[vv] = NEGINF;
#pragma unroll
    for (int jt = 0; jt < 4; ++jt) {
      int   jg = j0 + jt * 16 + ncol;
      float bj = biasS[jt * 16 + ncol];
#pragma unroll
      for (int vv = 0; vv < 8; ++vv) {
        float val = (st[jt][vv] + bj) * ATT_SCALE;
        if (jg > rrow + vv) val = NEGINF;  // causal mask
        st[jt][vv] = val;
        rowmax[vv] = fmaxf(rowmax[vv], val);
      }
    }
#pragma unroll
    for (int vv = 0; vv < 8; ++vv)
#pragma unroll
      for (int off = 1; off < 16; off <<= 1)
        rowmax[vv] = fmaxf(rowmax[vv], __shfl_xor(rowmax[vv], off, 16));
    float alpha[8];
#pragma unroll
    for (int vv = 0; vv < 8; ++vv) {
      float mnew = fmaxf(m_[vv], rowmax[vv]);
      alpha[vv] = __expf(m_[vv] - mnew);
      m_[vv] = mnew;
    }
    float rowsum[8];
#pragma unroll
    for (int vv = 0; vv < 8; ++vv) rowsum[vv] = 0.f;
#pragma unroll
    for (int jt = 0; jt < 4; ++jt)
#pragma unroll
      for (int vv = 0; vv < 8; ++vv) {
        float pv = __expf(st[jt][vv] - m_[vv]);
        st[jt][vv] = pv;
        rowsum[vv] += pv;
      }
#pragma unroll
    for (int vv = 0; vv < 8; ++vv) {
#pragma unroll
      for (int off = 1; off < 16; off <<= 1)
        rowsum[vv] += __shfl_xor(rowsum[vv], off, 16);
      l_[vv] = l_[vv] * alpha[vv] + rowsum[vv];
    }
#pragma unroll
    for (int nt = 0; nt < 4; ++nt)
#pragma unroll
      for (int vv = 0; vv < 8; ++vv) o[nt][vv] *= alpha[vv];

    // write P directly in A-fragment layout (wave-private LDS region)
    bf16* pw = &Ps[wave][0];
#pragma unroll
    for (int jt = 0; jt < 4; ++jt)
#pragma unroll
      for (int vv = 0; vv < 8; ++vv)
        pw[aswz(hi8 + vv, jt * 16 + ncol)] = (bf16)st[jt][vv];
    v16bf ap0 = *reinterpret_cast<const v16bf*>(pw + lane * 16);
    v16bf ap1 = *reinterpret_cast<const v16bf*>(pw + 512 + lane * 16);
    // O += P·V : B(k=j, n=d) = Vs[d][j] -> per-lane 16 contiguous j values
#pragma unroll
    for (int nt = 0; nt < 4; ++nt) {
      const bf16* vrow = &Vs[(nt * 16 + ncol) * 64 + kb];
      o[nt] = wmma_bf16(ap0, *reinterpret_cast<const v16bf*>(vrow), o[nt]);
      o[nt] = wmma_bf16(ap1, *reinterpret_cast<const v16bf*>(vrow + 32), o[nt]);
    }
    __syncthreads();
  }

  // epilogue: O /= l, store awv bf16 row-major (S,B,H*d)
#pragma unroll
  for (int vv = 0; vv < 8; ++vv) {
    float inv = 1.0f / l_[vv];
    int row = i0 + wave * 16 + hi8 + vv;
    size_t g = ((size_t)row * BSZ + b) * DM + coff + ncol;
#pragma unroll
    for (int nt = 0; nt < 4; ++nt) O[g + nt * 16] = (bf16)(o[nt][vv] * inv);
  }
}

// ---------------- host orchestration ----------------

extern "C" void kernel_launch(void* const* d_in, const int* in_sizes, int n_in,
                              void* d_out, int out_size, void* d_ws, size_t ws_size,
                              hipStream_t stream) {
  (void)in_sizes; (void)n_in; (void)out_size; (void)ws_size;
  const float* input_ = (const float*)d_in[0];
  const float* pos    = (const float*)d_in[1];
  const float* u      = (const float*)d_in[2];
  const float* v      = (const float*)d_in[3];
  const float* Wkv    = (const float*)d_in[4];
  const float* Wq     = (const float*)d_in[5];
  const float* Wp     = (const float*)d_in[6];
  const float* Wout   = (const float*)d_in[7];
  // d_in[8] (mask) unused: causal mask computed inline.

  char* ws = (char*)d_ws;
  size_t off = 0;
  auto carve = [&](size_t bytes) -> void* {
    void* p = ws + off;
    off = (off + bytes + 255) & ~(size_t)255;
    return p;
  };
  const size_t SBD = (size_t)SEQ * BSZ * DM;  // 8,388,608
  bf16*  Xbf   = (bf16*)carve(SBD * 2);
  bf16*  Pbf   = (bf16*)carve((size_t)SEQ * DM * 2);
  bf16*  Wkvt  = (bf16*)carve((size_t)DM * KVW * 2);  // [2048][1024]
  bf16*  Wqt   = (bf16*)carve((size_t)DM * DM * 2);   // [1024][1024]
  bf16*  Wpt   = (bf16*)carve((size_t)DM * DM * 2);
  bf16*  Woutt = (bf16*)carve((size_t)DM * DM * 2);
  float* kvf   = (float*)carve((size_t)SEQ * BSZ * KVW * 4);
  float* qf    = (float*)carve(SBD * 4);
  float* pf    = (float*)carve((size_t)SEQ * DM * 4);
  bf16*  qswz  = (bf16*)carve(SBD * 2);
  bf16*  kpb   = (bf16*)carve(SBD * 2);
  bf16*  vt    = (bf16*)carve(SBD * 2);
  float* biasb = (float*)carve((size_t)SEQ * BSZ * NH * 4);
  bf16*  awvb  = (bf16*)carve(SBD * 2);

  // bf16 conversions (A operands row-major, weights transposed to [N][K])
  f32_to_bf16_kernel<<<(int)(SBD / 256), 256, 0, stream>>>(input_, Xbf, (int)SBD);
  f32_to_bf16_kernel<<<SEQ * DM / 256, 256, 0, stream>>>(pos, Pbf, SEQ * DM);
  transpose_to_bf16_kernel<<<DM * KVW / 256, 256, 0, stream>>>(Wkv, Wkvt, KVW, DM);
  transpose_to_bf16_kernel<<<DM * DM / 256, 256, 0, stream>>>(Wq, Wqt, DM, DM);
  transpose_to_bf16_kernel<<<DM * DM / 256, 256, 0, stream>>>(Wp, Wpt, DM, DM);
  transpose_to_bf16_kernel<<<DM * DM / 256, 256, 0, stream>>>(Wout, Woutt, DM, DM);

  // projections
  gemm_bf16_kernel<<<dim3(KVW / 64, SEQ * BSZ / 64), 128, 0, stream>>>(
      Xbf, Wkvt, kvf, SEQ * BSZ, KVW, DM);
  gemm_bf16_kernel<<<dim3(DM / 64, SEQ * BSZ / 64), 128, 0, stream>>>(
      Xbf, Wqt, qf, SEQ * BSZ, DM, DM);
  gemm_bf16_kernel<<<dim3(DM / 64, SEQ / 64), 128, 0, stream>>>(
      Pbf, Wpt, pf, SEQ, DM, DM);

  // fold K' = k+p, swizzle q, transpose val, build per-j bias
  fold_kernel<<<(int)(SBD / 256), 256, 0, stream>>>(kvf, qf, pf, qswz, kpb, vt);
  bias_kernel<<<SEQ * BSZ * NH / 256, 256, 0, stream>>>(kvf, pf, u, v, biasb);

  // flash attention
  attn_kernel<<<dim3(SEQ / 64, BSZ * NH), 128, 0, stream>>>(qswz, kpb, vt, biasb, awvb);

  // output projection -> d_out (S,B,1024) f32
  gemm_bf16_kernel<<<dim3(DM / 64, SEQ * BSZ / 64), 128, 0, stream>>>(
      awvb, Woutt, (float*)d_out, SEQ * BSZ, DM, DM);
}
